// Weave_encoder_44641890075076
// MI455X (gfx1250) — compile-verified, matrix operand order checked by reading, not run on hardware
//
#include <hip/hip_runtime.h>
#include <stdint.h>

#define NN 50000
#define NE 800000

typedef __attribute__((ext_vector_type(16))) __bf16 v16bf;
typedef __attribute__((ext_vector_type(8)))  float  v8f;

union Frag {
  v16bf v;
  uint32_t u[8];
  uint4 q[2];
};

__device__ __forceinline__ uint16_t f2bf_u(float f) {
  union { float f; uint32_t u; } cv; cv.f = f;
  uint32_t u = cv.u;
  return (uint16_t)((u + 0x7FFFu + ((u >> 16) & 1u)) >> 16);  // RNE
}

__device__ __forceinline__ v8f wmma_bf16(const Frag& a, const Frag& b, v8f c) {
  return __builtin_amdgcn_wmma_f32_16x16x32_bf16(false, a.v, false, b.v,
                                                 (short)0, c, false, false);
}

// K index of the low element held in A/B VGPR v for this lane-half (ISA 16-bit layout)
__device__ __forceinline__ int kfrag(int v, int half) {
  return ((v >> 2) << 4) + (half << 3) + ((v & 3) << 1);
}

// A fragment from a bf16 row (global or LDS): lane's 8 dwords are two
// contiguous 16B spans at byte offsets half*16 and 32+half*16.
__device__ __forceinline__ void load_a_pair(Frag& a, const uint16_t* base, int half) {
  const char* bp = (const char*)base + half * 16;
  a.q[0] = *(const uint4*)(bp);
  a.q[1] = *(const uint4*)(bp + 32);
}

// A fragment from an f32 row (aggr): convert on the fly.
__device__ __forceinline__ void load_a_f32g(Frag& a, const float* base, int half) {
#pragma unroll
  for (int v = 0; v < 8; ++v) {
    int k0 = kfrag(v, half);
    float2 p = *(const float2*)(base + k0);
    a.u[v] = (uint32_t)f2bf_u(p.x) | ((uint32_t)f2bf_u(p.y) << 16);
  }
}

// B fragment from fragment-ordered packed weights: pkW[frag][lane][8] dwords.
__device__ __forceinline__ void load_b_frag(Frag& b, const uint32_t* __restrict__ pkW,
                                            int frag_idx, int lane) {
  const uint4* q = (const uint4*)(pkW + ((size_t)frag_idx * 32 + lane) * 8);
  b.q[0] = q[0];
  b.q[1] = q[1];
}

// ---------------- prep kernels ----------------

// Pack W (K x N f32, row-major) into fragment-ordered bf16 pairs:
// out[((ks*(N/16)+nt)*32 + lane)*8 + v] = {W[k][n], W[k+1][n]},
// k = ks*32 + kfrag(v, lane>>4), n = nt*16 + (lane&15).
__global__ __launch_bounds__(256) void pack_w_kernel(const float* __restrict__ W,
                                                     uint32_t* __restrict__ out,
                                                     int total, int N) {
  int i = blockIdx.x * blockDim.x + threadIdx.x;
  if (i >= total) return;
  int f = i >> 8;
  int rem = i & 255;
  int lane = rem >> 3;
  int v = rem & 7;
  int nt16 = N >> 4;
  int ks = f / nt16;
  int nt = f - ks * nt16;
  int half = lane >> 4;
  int n = nt * 16 + (lane & 15);
  int k0 = ks * 32 + kfrag(v, half);
  float lo = W[(size_t)k0 * N + n];
  float hi = W[(size_t)(k0 + 1) * N + n];
  out[i] = (uint32_t)f2bf_u(lo) | ((uint32_t)f2bf_u(hi) << 16);
}

__global__ __launch_bounds__(256) void cvt_bf16_kernel(const float* __restrict__ in,
                                                       uint16_t* __restrict__ out, size_t n) {
  size_t i = (size_t)blockIdx.x * blockDim.x + threadIdx.x;
  if (i < n) out[i] = f2bf_u(in[i]);
}

__global__ __launch_bounds__(256) void zero_kernel(float* __restrict__ p, size_t n) {
  size_t i = (size_t)blockIdx.x * blockDim.x + threadIdx.x;
  if (i < n) p[i] = 0.0f;
}

__global__ __launch_bounds__(256) void scatter_kernel(const float* __restrict__ e,
                                                      const int* __restrict__ dst,
                                                      float* __restrict__ aggr, size_t n) {
  size_t i = (size_t)blockIdx.x * blockDim.x + threadIdx.x;
  if (i >= n) return;
  int eidx = (int)(i >> 6);
  int c = (int)(i & 63);
  __hip_atomic_fetch_add(&aggr[(size_t)dst[eidx] * 64 + c], e[i],
                         __ATOMIC_RELAXED, __HIP_MEMORY_SCOPE_AGENT);
}

// ---------------- node kernel: x' = relu([relu(xW0+b0) | aggr] W1 + b1) ----------------

__global__ __launch_bounds__(128)
void node_kernel(const uint16_t* __restrict__ xb_in,   // NN x 128 bf16
                 const float* __restrict__ aggr,       // NN x 64 f32
                 const uint32_t* __restrict__ pkW0, const float* __restrict__ b0,
                 const uint32_t* __restrict__ pkW1, const float* __restrict__ b1,
                 float* __restrict__ x_out,            // NN x 128 f32
                 uint16_t* __restrict__ xb_out) {      // NN x 128 bf16
  __shared__ uint16_t lds_t0[4][16 * 64];
  const int lane = threadIdx.x & 31;
  const int wv = threadIdx.x >> 5;
  const int tile = blockIdx.x * 4 + wv;
  if (tile >= NN / 16) return;
  const int row = lane & 15, half = lane >> 4;
  const int rbase = tile * 16;
  const int ncol = row;

  // GEMM1: t0 = relu(x @ W0 + b0), K=128, N=64 (NT=4)
  v8f acc[4] = {{}, {}, {}, {}};
  const uint16_t* xrow = xb_in + (size_t)(rbase + row) * 128;
#pragma unroll
  for (int ks = 0; ks < 4; ++ks) {
    Frag a; load_a_pair(a, xrow + ks * 32, half);
#pragma unroll
    for (int nt = 0; nt < 4; ++nt) {
      Frag b; load_b_frag(b, pkW0, ks * 4 + nt, lane);
      acc[nt] = wmma_bf16(a, b, acc[nt]);
    }
  }
  uint16_t* t0 = lds_t0[wv];
#pragma unroll
  for (int nt = 0; nt < 4; ++nt) {
    int n = nt * 16 + ncol;
    float bias = b0[n];
#pragma unroll
    for (int r = 0; r < 8; ++r) {
      int m = r + half * 8;
      t0[m * 64 + n] = f2bf_u(fmaxf(acc[nt][r] + bias, 0.0f));
    }
  }
  asm volatile("s_wait_dscnt 0" ::: "memory");

  // GEMM2: out = relu([t0 | aggr] @ W1 + b1), K=128, N=128 (NT=8)
  v8f acc2[8] = {{}, {}, {}, {}, {}, {}, {}, {}};
  const float* arow = aggr + (size_t)(rbase + row) * 64;
  const uint16_t* t0row = t0 + row * 64;
#pragma unroll
  for (int ks = 0; ks < 4; ++ks) {
    Frag a;
    if (ks < 2) load_a_pair(a, t0row + ks * 32, half);
    else        load_a_f32g(a, arow + (ks - 2) * 32, half);
#pragma unroll
    for (int nt = 0; nt < 8; ++nt) {
      Frag b; load_b_frag(b, pkW1, ks * 8 + nt, lane);
      acc2[nt] = wmma_bf16(a, b, acc2[nt]);
    }
  }
#pragma unroll
  for (int nt = 0; nt < 8; ++nt) {
    int n = nt * 16 + ncol;
    float bias = b1[n];
#pragma unroll
    for (int r = 0; r < 8; ++r) {
      int m = r + half * 8;
      float v = fmaxf(acc2[nt][r] + bias, 0.0f);
      size_t o = (size_t)(rbase + m) * 128 + n;
      x_out[o] = v;
      xb_out[o] = f2bf_u(v);
    }
  }
}

// -------- edge kernel: e' = relu([relu(eW2+b2) | relu([x_i|x_j]W3+b3)] W4 + b4) --------

__global__ __launch_bounds__(128)
void edge_kernel(const uint16_t* __restrict__ eb_in,   // NE x 64 bf16
                 const uint16_t* __restrict__ xb_in,   // NN x 128 bf16
                 const int* __restrict__ src, const int* __restrict__ dst,
                 const uint32_t* __restrict__ pkW2, const float* __restrict__ b2,
                 const uint32_t* __restrict__ pkW3, const float* __restrict__ b3,
                 const uint32_t* __restrict__ pkW4, const float* __restrict__ b4,
                 float* __restrict__ e_out,            // NE x 64 f32
                 uint16_t* __restrict__ eb_out) {      // NE x 64 bf16
  __shared__ uint16_t lds_cat[4][16 * 128];
  const int lane = threadIdx.x & 31;
  const int wv = threadIdx.x >> 5;
  const int tile = blockIdx.x * 4 + wv;
  if (tile >= NE / 16) return;
  const int row = lane & 15, half = lane >> 4;
  const int rbase = tile * 16;
  const int ncol = row;

  const int er = rbase + row;
  const int di = dst[er];
  const int si = src[er];
  const uint16_t* xdi = xb_in + (size_t)di * 128;
  const uint16_t* xsi = xb_in + (size_t)si * 128;

  // GEMM1: cat2 = relu([x_i | x_j] @ W3 + b3), K=256, N=64 (NT=4)
  v8f c2[4] = {{}, {}, {}, {}};
#pragma unroll
  for (int ks = 0; ks < 8; ++ks) {
    const uint16_t* abase = (ks < 4) ? (xdi + ks * 32) : (xsi + (ks - 4) * 32);
    Frag a; load_a_pair(a, abase, half);
#pragma unroll
    for (int nt = 0; nt < 4; ++nt) {
      Frag b; load_b_frag(b, pkW3, ks * 4 + nt, lane);
      c2[nt] = wmma_bf16(a, b, c2[nt]);
    }
  }

  // GEMM2: cat1 = relu(e @ W2 + b2), K=64, N=64 (NT=4)
  v8f c1[4] = {{}, {}, {}, {}};
  const uint16_t* erow = eb_in + (size_t)er * 64;
#pragma unroll
  for (int ks = 0; ks < 2; ++ks) {
    Frag a; load_a_pair(a, erow + ks * 32, half);
#pragma unroll
    for (int nt = 0; nt < 4; ++nt) {
      Frag b; load_b_frag(b, pkW2, ks * 4 + nt, lane);
      c1[nt] = wmma_bf16(a, b, c1[nt]);
    }
  }

  // stage [cat1 | cat2] to LDS as bf16, 16 x 128
  uint16_t* cat = lds_cat[wv];
#pragma unroll
  for (int nt = 0; nt < 4; ++nt) {
    int n = nt * 16 + ncol;
    float bi1 = b2[n];
    float bi2 = b3[n];
#pragma unroll
    for (int r = 0; r < 8; ++r) {
      int m = r + half * 8;
      cat[m * 128 + n] = f2bf_u(fmaxf(c1[nt][r] + bi1, 0.0f));
      cat[m * 128 + 64 + n] = f2bf_u(fmaxf(c2[nt][r] + bi2, 0.0f));
    }
  }
  asm volatile("s_wait_dscnt 0" ::: "memory");

  // GEMM3: out = relu([cat1 | cat2] @ W4 + b4), K=128, N=64 (NT=4)
  v8f c3[4] = {{}, {}, {}, {}};
  const uint16_t* catrow = cat + row * 128;
#pragma unroll
  for (int ks = 0; ks < 4; ++ks) {
    Frag a; load_a_pair(a, catrow + ks * 32, half);
#pragma unroll
    for (int nt = 0; nt < 4; ++nt) {
      Frag b; load_b_frag(b, pkW4, ks * 4 + nt, lane);
      c3[nt] = wmma_bf16(a, b, c3[nt]);
    }
  }
#pragma unroll
  for (int nt = 0; nt < 4; ++nt) {
    int n = nt * 16 + ncol;
    float bias = b4[n];
#pragma unroll
    for (int r = 0; r < 8; ++r) {
      int m = r + half * 8;
      float v = fmaxf(c3[nt][r] + bias, 0.0f);
      size_t o = (size_t)(rbase + m) * 64 + n;
      e_out[o] = v;
      eb_out[o] = f2bf_u(v);
    }
  }
}

// ---------------- host ----------------

extern "C" void kernel_launch(void* const* d_in, const int* in_sizes, int n_in,
                              void* d_out, int out_size, void* d_ws, size_t ws_size,
                              hipStream_t stream) {
  const float* x0 = (const float*)d_in[0];
  const float* e0 = (const float*)d_in[1];
  const int* eidx = (const int*)d_in[2];
  const int* srcp = eidx;           // edge_index[0]
  const int* dstp = eidx + NE;      // edge_index[1]

  // Detect param flattening: sorted keys (W0..W4,b0..b4) vs interleaved (W0,b0,...)
  const bool interleaved = (in_sizes[4] == 64);
  const float* W[4][5];
  const float* B[4][5];
  for (int l = 0; l < 4; ++l)
    for (int j = 0; j < 5; ++j) {
      int base = 3 + l * 10;
      W[l][j] = (const float*)d_in[base + (interleaved ? 2 * j : j)];
      B[l][j] = (const float*)d_in[base + (interleaved ? 2 * j + 1 : 5 + j)];
    }

  // workspace carve-up
  char* p = (char*)d_ws;
  uint16_t* xb[2];
  xb[0] = (uint16_t*)p; p += (size_t)NN * 128 * 2;
  xb[1] = (uint16_t*)p; p += (size_t)NN * 128 * 2;
  uint16_t* eb[2];
  eb[0] = (uint16_t*)p; p += (size_t)NE * 64 * 2;
  eb[1] = (uint16_t*)p; p += (size_t)NE * 64 * 2;
  float* ef32 = (float*)p; p += (size_t)NE * 64 * 4;
  float* aggr = (float*)p; p += (size_t)NN * 64 * 4;
  static const int pkK[5] = {128, 128, 64, 256, 128};
  static const int pkN[5] = {64, 128, 64, 64, 64};
  uint32_t* pk[4][5];
  for (int l = 0; l < 4; ++l)
    for (int j = 0; j < 5; ++j) {
      pk[l][j] = (uint32_t*)p;
      p += (size_t)(pkK[j] / 2) * pkN[j] * 4;
    }

  // pack weights into fragment-ordered bf16 pairs
  for (int l = 0; l < 4; ++l)
    for (int j = 0; j < 5; ++j) {
      int tot = (pkK[j] / 2) * pkN[j];
      pack_w_kernel<<<(tot + 255) / 256, 256, 0, stream>>>(W[l][j], pk[l][j], tot, pkN[j]);
    }

  // bf16 copies of layer-0 activations
  {
    size_t nx = (size_t)NN * 128;
    cvt_bf16_kernel<<<(unsigned)((nx + 255) / 256), 256, 0, stream>>>(x0, xb[0], nx);
    size_t ne = (size_t)NE * 64;
    cvt_bf16_kernel<<<(unsigned)((ne + 255) / 256), 256, 0, stream>>>(e0, eb[0], ne);
  }

  const unsigned node_blocks = (NN / 16 + 3) / 4;   // 4 waves/block
  const unsigned edge_blocks = (NE / 16 + 3) / 4;

  int cx = 0, ce = 0;
  for (int l = 0; l < 4; ++l) {
    size_t na = (size_t)NN * 64;
    zero_kernel<<<(unsigned)((na + 255) / 256), 256, 0, stream>>>(aggr, na);

    const float* e_f32_in = (l == 0) ? e0 : ef32;
    size_t nsc = (size_t)NE * 64;
    scatter_kernel<<<(unsigned)((nsc + 255) / 256), 256, 0, stream>>>(e_f32_in, dstp, aggr, nsc);

    node_kernel<<<node_blocks, 128, 0, stream>>>(
        xb[cx], aggr, pk[l][0], B[l][0], pk[l][1], B[l][1],
        (float*)d_out, xb[cx ^ 1]);

    edge_kernel<<<edge_blocks, 128, 0, stream>>>(
        eb[ce], xb[cx], srcp, dstp,
        pk[l][2], B[l][2], pk[l][3], B[l][3], pk[l][4], B[l][4],
        ef32, eb[ce ^ 1]);

    cx ^= 1;
    ce ^= 1;
  }
  (void)n_in; (void)out_size; (void)ws_size;
}
